// MambaEncoder_3126736192210
// MI455X (gfx1250) — compile-verified, matrix operand order checked by reading, not run on hardware
//
#include <hip/hip_runtime.h>
#include <cstddef>
#include <cstdint>

// ---------------------------------------------------------------------------
// MambaEncoder for MI455X (gfx1250, wave32, WMMA + TDM).
// GEMMs: LDS-blocked, TDM (tensor_load_to_lds) double-buffered staging,
// 8 waves/block each computing a 32x32 C macro-tile with 2x2
// v_wmma_f32_16x16x32_bf16 accumulators. fp32 accumulate, bf16 operands.
// ---------------------------------------------------------------------------

typedef __bf16 bf16_t;
typedef __attribute__((ext_vector_type(16))) __bf16 v16bf;
typedef __attribute__((ext_vector_type(8)))  __bf16 v8bf;
typedef __attribute__((ext_vector_type(8)))  float  v8f;
typedef __attribute__((ext_vector_type(4)))  unsigned int v4u;
typedef __attribute__((ext_vector_type(8)))  int v8i;
typedef __attribute__((ext_vector_type(4)))  int v4i;

static constexpr int L        = 4096;
static constexpr int D_MODEL  = 1024;
static constexpr int D_INNER  = 2048;
static constexpr int D_STATE  = 16;
static constexpr int D_CONV   = 4;
static constexpr int DT_RANK  = 64;
static constexpr int XD       = DT_RANK + 2 * D_STATE;   // 96
static constexpr int E_HALF   = 512;

// GEMM blocking
static constexpr int BM = 64;     // block rows of C
static constexpr int BN = 128;    // block cols of C
static constexpr int BK = 32;     // K step (one WMMA depth)
static constexpr int LDS_PITCH = 40;   // elements per staged row (32 + 8 pad -> 80B pitch)

#if defined(__HIP_DEVICE_COMPILE__) && defined(__has_builtin)
# if __has_builtin(__builtin_amdgcn_tensor_load_to_lds) && __has_builtin(__builtin_amdgcn_s_wait_tensorcnt)
#  define USE_TDM 1
# endif
#endif

__device__ __forceinline__ bf16_t f2bf(float f) {
  unsigned u = __builtin_bit_cast(unsigned, f);
  unsigned r = (u + 0x7FFFu + ((u >> 16) & 1u)) >> 16;   // RNE
  unsigned short s = (unsigned short)r;
  return __builtin_bit_cast(bf16_t, s);
}

// ---------------------------------------------------------------------------
// TDM 2D tile load: rows x 32 bf16 elements, row-padded into LDS (pitch 80B).
// D# per CDNA5 ISA §8: group0 {count=1, lds_addr, global_addr, type=2},
// group1 {data_size=2B, pad_enable, pad_interval=3 (64B), pad_amount=3 (16B),
//         tensor_dim0=K, tensor_dim1=rows_total, tile_dim0=32, tile_dim1=rows,
//         tensor_dim0_stride=K}.
// ---------------------------------------------------------------------------
__device__ __forceinline__ void tdm_stage_2d(const bf16_t* gsrc, unsigned lds_addr,
                                             int K, int rows_total, int tile_rows)
{
#if defined(USE_TDM)
  unsigned long long ga = (unsigned long long)(const void*)gsrc;
  v4u g0 = { 1u,                                     // count=1
             lds_addr,
             (unsigned)(ga & 0xFFFFFFFFull),
             (unsigned)((ga >> 32) & 0x01FFFFFFull) | 0x80000000u };  // type=2
  unsigned uK = (unsigned)K, uR = (unsigned)rows_total;
  v8i g1 = { (int)(0x00010000u | (1u << 20) | (3u << 22) | (3u << 25)),
             (int)((uK & 0xFFFFu) << 16),
             (int)((uK >> 16) | ((uR & 0xFFFFu) << 16)),
             (int)((uR >> 16) | (32u << 16)),
             (int)(unsigned)tile_rows,
             (int)uK,                                // tensor_dim0_stride[31:0]
             0, 0 };
  v4i z4 = { 0, 0, 0, 0 };
# if __clang_major__ >= 23
  v8i z8 = { 0, 0, 0, 0, 0, 0, 0, 0 };
  __builtin_amdgcn_tensor_load_to_lds(g0, g1, z4, z4, z8, 0);
# else
  __builtin_amdgcn_tensor_load_to_lds(g0, g1, z4, z4, 0);
# endif
#else
  (void)gsrc; (void)lds_addr; (void)K; (void)rows_total; (void)tile_rows;
#endif
}

// ---------------------------------------------------------------------------
// LDS-blocked WMMA GEMM: C(fp32, MxN, ldc) = alpha * A(bf16 MxK) W(bf16 NxK)^T
//                                            + beta * C
// grid = (ceil(N/BN), M/BM), block = 256 (8 waves). K % 32 == 0, M % 64 == 0.
// Waves: 2 (M) x 4 (N); each computes 32x32 via 2x2 WMMA tiles.
// ---------------------------------------------------------------------------
__global__ __launch_bounds__(256)
void k_gemm_bf16_wmma(const bf16_t* __restrict__ A, const bf16_t* __restrict__ W,
                      float* __restrict__ C, int M, int N, int K, int ldc,
                      float alpha, float beta)
{
  __shared__ bf16_t lA[2][BM * LDS_PITCH];   // 2 x 5120 B
  __shared__ bf16_t lB[2][BN * LDS_PITCH];   // 2 x 10240 B

  const int lane = threadIdx.x & 31;
  const int wave = threadIdx.x >> 5;
  const int half = lane >> 4;
  const int row  = lane & 15;
  const int wm   = (wave >> 2) * 32;          // wave M offset within block tile
  const int wn   = (wave & 3) * 32;           // wave N offset within block tile
  const int m0   = blockIdx.y * BM;
  const int n0   = blockIdx.x * BN;
  const int nk   = K / BK;

  const bf16_t* gA = A + (size_t)m0 * K;
  const bf16_t* gW = W + (size_t)n0 * K;
  const int rowsB_total = N - n0;             // TDM OOB rows read as zero

  // ---- stage one K-slice (both tiles) into buffer `buf` at k-offset k0 ----
  auto stage = [&](int buf, int k0) {
#if defined(USE_TDM)
    if (threadIdx.x < 32) {                   // wave 0 issues both DMAs
      tdm_stage_2d(gA + k0, (unsigned)(uintptr_t)(void*)&lA[buf][0], K, M - m0, BM);
      tdm_stage_2d(gW + k0, (unsigned)(uintptr_t)(void*)&lB[buf][0], K, rowsB_total, BN);
    }
#else
    for (int e = threadIdx.x; e < BM * BK; e += 256) {
      int r = e >> 5, c = e & 31;
      lA[buf][r * LDS_PITCH + c] = gA[(size_t)r * K + k0 + c];
    }
    for (int e = threadIdx.x; e < BN * BK; e += 256) {
      int r = e >> 5, c = e & 31;
      bf16_t v = (r < rowsB_total) ? gW[(size_t)r * K + k0 + c] : f2bf(0.0f);
      lB[buf][r * LDS_PITCH + c] = v;
    }
#endif
  };

  v8f acc00 = {}, acc01 = {}, acc10 = {}, acc11 = {};

  stage(0, 0);
  for (int ki = 0; ki < nk; ++ki) {
    const int buf = ki & 1;
    const bool more = (ki + 1) < nk;
    if (more) stage(buf ^ 1, (ki + 1) * BK);
#if defined(USE_TDM)
    if (threadIdx.x < 32) {
      if (more) __builtin_amdgcn_s_wait_tensorcnt(2);  // current buffer's 2 done
      else      __builtin_amdgcn_s_wait_tensorcnt(0);
    }
#endif
    __syncthreads();

    // fragment pointers (ISA 7.12.2 layout: two 8-half runs at +8h and +16+8h)
    const bf16_t* a0 = &lA[buf][(wm + row) * LDS_PITCH + 8 * half];
    const bf16_t* a1 = &lA[buf][(wm + 16 + row) * LDS_PITCH + 8 * half];
    const bf16_t* b0 = &lB[buf][(wn + row) * LDS_PITCH + 8 * half];
    const bf16_t* b1 = &lB[buf][(wn + 16 + row) * LDS_PITCH + 8 * half];

    v8bf a0l = *(const v8bf*)a0, a0h = *(const v8bf*)(a0 + 16);
    v8bf a1l = *(const v8bf*)a1, a1h = *(const v8bf*)(a1 + 16);
    v8bf b0l = *(const v8bf*)b0, b0h = *(const v8bf*)(b0 + 16);
    v8bf b1l = *(const v8bf*)b1, b1h = *(const v8bf*)(b1 + 16);
    v16bf fa0, fa1, fb0, fb1;
#pragma unroll
    for (int i = 0; i < 8; ++i) {
      fa0[i] = a0l[i]; fa0[i + 8] = a0h[i];
      fa1[i] = a1l[i]; fa1[i + 8] = a1h[i];
      fb0[i] = b0l[i]; fb0[i + 8] = b0h[i];
      fb1[i] = b1l[i]; fb1[i + 8] = b1h[i];
    }
    acc00 = __builtin_amdgcn_wmma_f32_16x16x32_bf16(false, fa0, false, fb0, (short)0, acc00, false, false);
    acc01 = __builtin_amdgcn_wmma_f32_16x16x32_bf16(false, fa0, false, fb1, (short)0, acc01, false, false);
    acc10 = __builtin_amdgcn_wmma_f32_16x16x32_bf16(false, fa1, false, fb0, (short)0, acc10, false, false);
    acc11 = __builtin_amdgcn_wmma_f32_16x16x32_bf16(false, fa1, false, fb1, (short)0, acc11, false, false);

    __syncthreads();   // protect buffer reuse before next stage()
  }

  // ---- epilogue: D layout lane l -> col n=l%16, VGPR v -> row v + 8*(l/16) ----
  v8f accs[4] = { acc00, acc01, acc10, acc11 };
#pragma unroll
  for (int i = 0; i < 2; ++i) {
#pragma unroll
    for (int j = 0; j < 2; ++j) {
      int ncol0 = n0 + wn + j * 16;
      if (ncol0 >= N) continue;               // tile-uniform guard
      v8f a = accs[i * 2 + j];
      float* cp = C + (size_t)(m0 + wm + i * 16 + half * 8) * ldc + ncol0 + row;
#pragma unroll
      for (int v = 0; v < 8; ++v) {
        size_t idx = (size_t)v * ldc;
        float r = alpha * a[v];
        if (beta != 0.0f) r += beta * cp[idx];
        cp[idx] = r;
      }
    }
  }
}

// --------------------------- elementwise kernels ---------------------------

__global__ __launch_bounds__(256)
void k_concat_tokens(const float* __restrict__ img, const float* __restrict__ txt,
                     bf16_t* __restrict__ tok)
{
  int idx = blockIdx.x * blockDim.x + threadIdx.x;
  if (idx >= L * D_MODEL) return;
  int r = idx / D_MODEL, c = idx % D_MODEL;
  float v = (c < E_HALF) ? img[(size_t)r * E_HALF + c]
                         : txt[(size_t)r * E_HALF + (c - E_HALF)];
  tok[idx] = f2bf(v);
}

__global__ __launch_bounds__(256)
void k_f32_to_bf16(const float* __restrict__ src, bf16_t* __restrict__ dst, int n)
{
  int idx = blockIdx.x * blockDim.x + threadIdx.x;
  if (idx < n) dst[idx] = f2bf(src[idx]);
}

__global__ __launch_bounds__(256)
void k_conv_silu(const float* __restrict__ xz, const float* __restrict__ cw,
                 const float* __restrict__ cb, float* __restrict__ u,
                 bf16_t* __restrict__ ub)
{
  int idx = blockIdx.x * blockDim.x + threadIdx.x;
  if (idx >= L * D_INNER) return;
  int t = idx / D_INNER, d = idx % D_INNER;
  float acc = cb[d];
#pragma unroll
  for (int k = 0; k < D_CONV; ++k) {
    int tt = t + k - (D_CONV - 1);
    if (tt >= 0) acc += xz[(size_t)tt * (2 * D_INNER) + d] * cw[d * D_CONV + k];
  }
  float sil = acc / (1.0f + __expf(-acc));
  u[idx] = sil;
  ub[idx] = f2bf(sil);
}

__global__ __launch_bounds__(256)
void k_dt_cvt(const float* __restrict__ xdbl, bf16_t* __restrict__ dt)
{
  int idx = blockIdx.x * blockDim.x + threadIdx.x;
  if (idx >= L * DT_RANK) return;
  int r = idx >> 6, c = idx & 63;
  dt[idx] = f2bf(xdbl[(size_t)r * XD + c]);
}

__global__ __launch_bounds__(256)
void k_softplus_bias(float* __restrict__ delta, const float* __restrict__ bias)
{
  int idx = blockIdx.x * blockDim.x + threadIdx.x;
  if (idx >= L * D_INNER) return;
  float x = delta[idx] + bias[idx % D_INNER];
  delta[idx] = (x > 20.0f) ? x : log1pf(__expf(x));
}

__global__ __launch_bounds__(256)
void k_scan(const float* __restrict__ delta, const float* __restrict__ xdbl,
            const float* __restrict__ u, const float* __restrict__ xz,
            const float* __restrict__ A_log, const float* __restrict__ Dp,
            bf16_t* __restrict__ y)
{
  int tid = blockIdx.x * blockDim.x + threadIdx.x;
  int d = tid >> 4;
  int s = tid & 15;
  float A  = -__expf(A_log[d * D_STATE + s]);
  float Dd = Dp[d];
  float h = 0.0f;
  for (int t = 0; t < L; ++t) {
    float dt = delta[(size_t)t * D_INNER + d];
    float ut = u[(size_t)t * D_INNER + d];
    float Bt = xdbl[(size_t)t * XD + DT_RANK + s];
    float Ct = xdbl[(size_t)t * XD + DT_RANK + D_STATE + s];
    h = __expf(dt * A) * h + (dt * ut) * Bt;
    float p = h * Ct;
    p += __shfl_xor(p, 1);
    p += __shfl_xor(p, 2);
    p += __shfl_xor(p, 4);
    p += __shfl_xor(p, 8);
    if (s == 0) {
      float z = xz[(size_t)t * (2 * D_INNER) + D_INNER + d];
      float sz = z / (1.0f + __expf(-z));
      y[(size_t)t * D_INNER + d] = f2bf((p + ut * Dd) * sz);
    }
  }
}

__global__ __launch_bounds__(128)
void k_rowsum(const float* __restrict__ src, float* __restrict__ dst, int width)
{
  __shared__ float red[128];
  int r = blockIdx.x;
  float s = 0.0f;
  for (int c = threadIdx.x; c < width; c += 128) s += src[(size_t)r * width + c];
  red[threadIdx.x] = s;
  __syncthreads();
  for (int w = 64; w > 0; w >>= 1) {
    if (threadIdx.x < w) red[threadIdx.x] += red[threadIdx.x + w];
    __syncthreads();
  }
  if (threadIdx.x == 0) dst[r] = red[0];
}

__global__ __launch_bounds__(256)
void k_sm(const float* __restrict__ i1, const float* __restrict__ t1,
          bf16_t* __restrict__ sm)
{
  int idx = blockIdx.x * blockDim.x + threadIdx.x;
  if (idx >= L * L) return;
  int i = idx >> 12, j = idx & (L - 1);
  float x = i1[i] * i1[j] - t1[i] * t1[j];
  float sg = 1.0f / (1.0f + __expf(-x));
  float th = tanhf(x);
  sm[idx] = f2bf((1.0f - th * th) * sg * (1.0f - sg));
}

__global__ __launch_bounds__(256)
void k_transpose_cvt(const float* __restrict__ src, bf16_t* __restrict__ dst)
{
  int idx = blockIdx.x * blockDim.x + threadIdx.x;
  if (idx >= L * E_HALF) return;
  int j = idx / L, k = idx % L;
  dst[idx] = f2bf(src[(size_t)k * E_HALF + j]);
}

__global__ __launch_bounds__(256)
void k_normalize_out(const float* __restrict__ res, float* __restrict__ out)
{
  __shared__ float red[256];
  int r = blockIdx.x;
  const float* rp = res + (size_t)r * D_MODEL;
  float s = 0.0f;
  for (int c = threadIdx.x; c < D_MODEL; c += 256) { float v = rp[c]; s += v * v; }
  red[threadIdx.x] = s;
  __syncthreads();
  for (int w = 128; w > 0; w >>= 1) {
    if (threadIdx.x < w) red[threadIdx.x] += red[threadIdx.x + w];
    __syncthreads();
  }
  float inv = 1.0f / fmaxf(sqrtf(red[0]), 1e-12f);
  for (int c = threadIdx.x; c < D_MODEL; c += 256) {
    float v = rp[c] * inv;
    if (c < E_HALF) out[(size_t)r * E_HALF + c] = v;
    else            out[(size_t)L * E_HALF + (size_t)r * E_HALF + (c - E_HALF)] = v;
  }
}

// ------------------------------- host side ---------------------------------

static inline void launch_gemm(const bf16_t* A, const bf16_t* W, float* C,
                               int M, int N, int K, int ldc, float alpha,
                               float beta, hipStream_t s)
{
  dim3 grid((N + BN - 1) / BN, M / BM);
  k_gemm_bf16_wmma<<<grid, 256, 0, s>>>(A, W, C, M, N, K, ldc, alpha, beta);
}

// workspace layout (bytes)
static constexpr size_t OFF_XZ    = 0;                                        // 64 MB; reused by sm bf16 after scan
static constexpr size_t OFF_TOK   = OFF_XZ + (size_t)L * 2 * D_INNER * 4;     // 8 MB tokens bf16 (GEMM1 only)
static constexpr size_t OFF_DT    = OFF_TOK;                                  // reuse: dt bf16
static constexpr size_t OFF_I1    = OFF_TOK + (size_t)L * DT_RANK * 2;
static constexpr size_t OFF_T1    = OFF_I1 + (size_t)L * 4;
static constexpr size_t OFF_WBUF  = OFF_TOK + (size_t)L * D_MODEL * 2;        // 8 MB rotating weight buffer
static constexpr size_t OFF_U     = OFF_WBUF + (size_t)(2 * D_INNER) * D_MODEL * 2;
static constexpr size_t OFF_UB    = OFF_U + (size_t)L * D_INNER * 4;
static constexpr size_t OFF_DELTA = OFF_UB + (size_t)L * D_INNER * 2;
static constexpr size_t OFF_Y     = OFF_DELTA + (size_t)L * D_INNER * 4;
static constexpr size_t OFF_RES   = OFF_Y + (size_t)L * D_INNER * 2;
static constexpr size_t OFF_XDBL  = OFF_RES + (size_t)L * D_MODEL * 4;

extern "C" void kernel_launch(void* const* d_in, const int* in_sizes, int n_in,
                              void* d_out, int out_size, void* d_ws, size_t ws_size,
                              hipStream_t stream)
{
  (void)in_sizes; (void)n_in; (void)out_size; (void)ws_size;
  const float* img       = (const float*)d_in[0];
  const float* txt       = (const float*)d_in[1];
  const float* in_proj_w = (const float*)d_in[2];
  const float* conv_w    = (const float*)d_in[3];
  const float* conv_b    = (const float*)d_in[4];
  const float* x_proj_w  = (const float*)d_in[5];
  const float* dt_proj_w = (const float*)d_in[6];
  const float* dt_proj_b = (const float*)d_in[7];
  const float* A_log     = (const float*)d_in[8];
  const float* Dp        = (const float*)d_in[9];
  const float* out_proj  = (const float*)d_in[10];
  float* out = (float*)d_out;

  char* ws = (char*)d_ws;
  float*  xz    = (float*)(ws + OFF_XZ);
  bf16_t* tok   = (bf16_t*)(ws + OFF_TOK);
  bf16_t* dtb   = (bf16_t*)(ws + OFF_DT);
  float*  i1    = (float*)(ws + OFF_I1);
  float*  t1    = (float*)(ws + OFF_T1);
  bf16_t* wbuf  = (bf16_t*)(ws + OFF_WBUF);
  float*  u     = (float*)(ws + OFF_U);
  bf16_t* ub    = (bf16_t*)(ws + OFF_UB);
  float*  delta = (float*)(ws + OFF_DELTA);
  bf16_t* y     = (bf16_t*)(ws + OFF_Y);
  float*  res   = (float*)(ws + OFF_RES);
  float*  xdbl  = (float*)(ws + OFF_XDBL);
  bf16_t* sm    = (bf16_t*)(ws + OFF_XZ);          // reuses xz region after scan

  auto blks = [](int n) { return (n + 255) / 256; };

  // 1) xz = tokens @ in_proj^T
  k_concat_tokens<<<blks(L * D_MODEL), 256, 0, stream>>>(img, txt, tok);
  k_f32_to_bf16<<<blks(2 * D_INNER * D_MODEL), 256, 0, stream>>>(in_proj_w, wbuf, 2 * D_INNER * D_MODEL);
  launch_gemm(tok, wbuf, xz, L, 2 * D_INNER, D_MODEL, 2 * D_INNER, 1.0f, 0.0f, stream);

  // 2) causal conv + silu
  k_conv_silu<<<blks(L * D_INNER), 256, 0, stream>>>(xz, conv_w, conv_b, u, ub);

  // 3) x_dbl = u @ x_proj^T   (N = 96)
  k_f32_to_bf16<<<blks(XD * D_INNER), 256, 0, stream>>>(x_proj_w, wbuf, XD * D_INNER);
  launch_gemm(ub, wbuf, xdbl, L, XD, D_INNER, XD, 1.0f, 0.0f, stream);

  // 4) delta = softplus(dt @ dt_proj^T + b)
  k_dt_cvt<<<blks(L * DT_RANK), 256, 0, stream>>>(xdbl, dtb);
  k_f32_to_bf16<<<blks(D_INNER * DT_RANK), 256, 0, stream>>>(dt_proj_w, wbuf, D_INNER * DT_RANK);
  launch_gemm(dtb, wbuf, delta, L, D_INNER, DT_RANK, D_INNER, 1.0f, 0.0f, stream);
  k_softplus_bias<<<blks(L * D_INNER), 256, 0, stream>>>(delta, dt_proj_b);

  // 5) selective scan + z-gate -> y (bf16)
  k_scan<<<(D_INNER * D_STATE) / 256, 256, 0, stream>>>(delta, xdbl, u, xz, A_log, Dp, y);

  // 6) res = y @ out_proj^T
  k_f32_to_bf16<<<blks(D_MODEL * D_INNER), 256, 0, stream>>>(out_proj, wbuf, D_MODEL * D_INNER);
  launch_gemm(y, wbuf, res, L, D_MODEL, D_INNER, D_MODEL, 1.0f, 0.0f, stream);

  // 7) feature enhance: res[:, :512] += 0.1*sm@img ; res[:, 512:] += 0.1*sm@txt
  k_rowsum<<<L, 128, 0, stream>>>(img, i1, E_HALF);
  k_rowsum<<<L, 128, 0, stream>>>(txt, t1, E_HALF);
  k_sm<<<blks(L * L), 256, 0, stream>>>(i1, t1, sm);
  k_transpose_cvt<<<blks(L * E_HALF), 256, 0, stream>>>(img, wbuf);
  launch_gemm(sm, wbuf, res, L, E_HALF, L, D_MODEL, 0.1f, 1.0f, stream);
  k_transpose_cvt<<<blks(L * E_HALF), 256, 0, stream>>>(txt, wbuf);
  launch_gemm(sm, wbuf, res + E_HALF, L, E_HALF, L, D_MODEL, 0.1f, 1.0f, stream);

  // 8) normalize rows, split into tuple layout
  k_normalize_out<<<L, 256, 0, stream>>>(res, out);
}
